// Transformer_32495722561810
// MI455X (gfx1250) — compile-verified, matrix operand order checked by reading, not run on hardware
//
#include <hip/hip_runtime.h>

// ---------------------------------------------------------------------------
// CDNA5 (gfx1250) fused transformer block:
//   qkv GEMM (bf16 WMMA) -> 128-token block-causal attention (bf16 WMMA,
//   register softmax) -> output projection GEMM (bf16 WMMA, fp32 out)
// ---------------------------------------------------------------------------

typedef __attribute__((ext_vector_type(16))) __bf16 v16bf;
typedef __attribute__((ext_vector_type(8)))  float  v8f;

union Frag { uint4 u[2]; v16bf v; };

__device__ __forceinline__ v16bf load_frag(const __bf16* p) {
  // 16 bf16 elements in WMMA A/B lane order: elems 0..7 at p, elems 8..15 at p+16
  Frag f;
  f.u[0] = *(const uint4*)(p);
  f.u[1] = *(const uint4*)(p + 16);
  return f.v;
}

// ---------------------------------------------------------------------------
// fp32 -> bf16 elementwise convert (x)
// ---------------------------------------------------------------------------
__global__ __launch_bounds__(256) void cvt_bf16(const float* __restrict__ in,
                                                __bf16* __restrict__ out,
                                                size_t n) {
  size_t i = ((size_t)blockIdx.x * blockDim.x + threadIdx.x) * 4;
  if (i + 3 < n) {
    float4 f = *(const float4*)(in + i);
    out[i + 0] = (__bf16)f.x;
    out[i + 1] = (__bf16)f.y;
    out[i + 2] = (__bf16)f.z;
    out[i + 3] = (__bf16)f.w;
  }
}

// ---------------------------------------------------------------------------
// fp32 (R x Ccols) -> bf16 transposed (Ccols x R), LDS tiled 32x32
// ---------------------------------------------------------------------------
__global__ __launch_bounds__(256) void transpose_cvt(const float* __restrict__ in,
                                                     __bf16* __restrict__ out,
                                                     int R, int Ccols) {
  __shared__ float tile[32][33];
  int c0 = blockIdx.x * 32;
  int r0 = blockIdx.y * 32;
  int tx = threadIdx.x;           // 0..31
  int ty = threadIdx.y;           // 0..7
#pragma unroll
  for (int i = 0; i < 32; i += 8)
    tile[ty + i][tx] = in[(size_t)(r0 + ty + i) * Ccols + c0 + tx];
  __syncthreads();
#pragma unroll
  for (int i = 0; i < 32; i += 8)
    out[(size_t)(c0 + ty + i) * R + r0 + tx] = (__bf16)tile[tx][ty + i];
}

// ---------------------------------------------------------------------------
// Tiled bf16 GEMM:  C(MxN) = A(MxK, row-major) * B, with B given transposed as
// BT(NxK, row-major).  Block tile 128x128, 8 wave32s, wave tile 32x64
// (2x4 grid of v_wmma_f32_16x16x32_bf16).  LDS row stride 40 bf16 (80B) keeps
// ds_load_b128 fragment reads bank-conflict free.
// ---------------------------------------------------------------------------
template <bool OUT_BF16>
__global__ __launch_bounds__(256) void gemm_bf16_kernel(const __bf16* __restrict__ A,
                                                        const __bf16* __restrict__ BT,
                                                        void* __restrict__ Cout,
                                                        int M, int N, int K) {
  __shared__ __align__(16) __bf16 sA[128 * 40];
  __shared__ __align__(16) __bf16 sB[128 * 40];

  const int tid   = threadIdx.x;
  const int wave  = tid >> 5;
  const int lane  = tid & 31;
  const int l15   = lane & 15;
  const int khalf = (lane & 16) ? 8 : 0;   // K sub-offset per WMMA 16-bit layout
  const int hAdd  = (lane & 16) ? 8 : 0;   // C-layout row offset for upper half
  const int waveM = wave >> 1;             // 0..3
  const int waveN = wave & 1;              // 0..1

  const int bm = blockIdx.y * 128;
  const int bn = blockIdx.x * 128;

  v8f zero = {};
  v8f acc[2][4];
#pragma unroll
  for (int mt = 0; mt < 2; ++mt)
#pragma unroll
    for (int nt = 0; nt < 4; ++nt) acc[mt][nt] = zero;

  // Staging assignment: thread t loads 16 bf16 of one row-slab (A and BT each)
  const int srow = tid >> 1;
  const int scol = (tid & 1) * 16;
  const __bf16* ag = A  + (size_t)(bm + srow) * K + scol;
  const __bf16* bg = BT + (size_t)(bn + srow) * K + scol;
  __bf16* sAp = &sA[srow * 40 + scol];
  __bf16* sBp = &sB[srow * 40 + scol];

  for (int kk = 0; kk < K; kk += 32) {
    uint4 a0 = *(const uint4*)(ag + kk);
    uint4 a1 = *(const uint4*)(ag + kk + 8);
    uint4 b0 = *(const uint4*)(bg + kk);
    uint4 b1 = *(const uint4*)(bg + kk + 8);
    *(uint4*)(sAp)     = a0;
    *(uint4*)(sAp + 8) = a1;
    *(uint4*)(sBp)     = b0;
    *(uint4*)(sBp + 8) = b1;
    __syncthreads();

    Frag af[2], bf_[4];
#pragma unroll
    for (int mt = 0; mt < 2; ++mt) {
      const __bf16* p = &sA[(waveM * 32 + mt * 16 + l15) * 40 + khalf];
      af[mt].u[0] = *(const uint4*)p;
      af[mt].u[1] = *(const uint4*)(p + 16);
    }
#pragma unroll
    for (int nt = 0; nt < 4; ++nt) {
      const __bf16* p = &sB[(waveN * 64 + nt * 16 + l15) * 40 + khalf];
      bf_[nt].u[0] = *(const uint4*)p;
      bf_[nt].u[1] = *(const uint4*)(p + 16);
    }
#pragma unroll
    for (int mt = 0; mt < 2; ++mt)
#pragma unroll
      for (int nt = 0; nt < 4; ++nt)
        acc[mt][nt] = __builtin_amdgcn_wmma_f32_16x16x32_bf16(
            false, af[mt].v, false, bf_[nt].v, (short)0, acc[mt][nt], false, false);
    __syncthreads();
  }

  // Store C from WMMA C-layout: VGPR e, lanes 0-15 -> row e; lanes 16-31 -> row e+8
#pragma unroll
  for (int mt = 0; mt < 2; ++mt) {
#pragma unroll
    for (int nt = 0; nt < 4; ++nt) {
      const int rbase = bm + waveM * 32 + mt * 16 + hAdd;
      const int col   = bn + waveN * 64 + nt * 16 + l15;
      if (OUT_BF16) {
        __bf16* C = (__bf16*)Cout;
#pragma unroll
        for (int e = 0; e < 8; ++e)
          C[(size_t)(rbase + e) * N + col] = (__bf16)acc[mt][nt][e];
      } else {
        float* C = (float*)Cout;
#pragma unroll
        for (int e = 0; e < 8; ++e)
          C[(size_t)(rbase + e) * N + col] = acc[mt][nt][e];
      }
    }
  }
}

// ---------------------------------------------------------------------------
// Fused block-causal attention.  One workgroup (8 wave32s) per (b, blk, h).
// q,k,v are 128x128 bf16 tiles inside the qkv buffer (32768 x 6144 bf16).
// scores kept in registers; softmax via xor-shuffles over 16-lane halves.
// LDS row stride 136 bf16 (272B) -> conflict-free ds_load_b128.
// ---------------------------------------------------------------------------
__global__ __launch_bounds__(256) void attn_kernel(const __bf16* __restrict__ qkv,
                                                   __bf16* __restrict__ y) {
  __shared__ __align__(16) __bf16 sK[128 * 136];       // K tile, row-major (j, d)
  __shared__ __align__(16) __bf16 sVT[128 * 136];      // V^T tile (d, j)
  __shared__ __align__(16) __bf16 sAtt[8 * 16 * 136];  // per-wave att strips

  const int bid = blockIdx.x;
  const int h  = bid & 15;
  const int nb = (bid >> 4) & 63;
  const int b  = bid >> 10;
  const size_t tok0 = (size_t)b * 8192 + (size_t)nb * 128;

  const __bf16* qbase = qkv + tok0 * 6144 + h * 128;
  const __bf16* kbase = qbase + 2048;
  const __bf16* vbase = qbase + 4096;

  const int tid = threadIdx.x;

  // ---- stage K (direct) and V (transposed) into LDS ----
  {
    const int row = tid >> 1;             // token j = 0..127
    const int seg = (tid & 1) * 64;       // d segment
    const __bf16* ks = kbase + (size_t)row * 6144 + seg;
    const __bf16* vs = vbase + (size_t)row * 6144 + seg;
#pragma unroll
    for (int i = 0; i < 8; ++i) {
      uint4 kv = *(const uint4*)(ks + i * 8);
      *(uint4*)(&sK[row * 136 + seg + i * 8]) = kv;
      uint4 vv = *(const uint4*)(vs + i * 8);
      const __bf16* ve = (const __bf16*)&vv;
#pragma unroll
      for (int j = 0; j < 8; ++j)
        sVT[(seg + i * 8 + j) * 136 + row] = ve[j];
    }
  }
  __syncthreads();

  const int wave  = tid >> 5;
  const int lane  = tid & 31;
  const int l15   = lane & 15;
  const int khalf = (lane & 16) ? 8 : 0;
  const int hAdd  = (lane & 16) ? 8 : 0;
  const int m0    = wave * 16;            // this wave's 16 query rows

  // ---- Q fragments straight from global ----
  Frag qf[4];
  {
    const __bf16* qrow = qbase + (size_t)(m0 + l15) * 6144;
#pragma unroll
    for (int kk = 0; kk < 4; ++kk) {
      qf[kk].u[0] = *(const uint4*)(qrow + kk * 32 + khalf);
      qf[kk].u[1] = *(const uint4*)(qrow + kk * 32 + khalf + 16);
    }
  }

  // ---- scores = q @ k^T  (8 N-tiles x 4 K-steps of WMMA) ----
  v8f zero = {};
  v8f s[8];
#pragma unroll
  for (int nt = 0; nt < 8; ++nt) {
    v8f a = zero;
#pragma unroll
    for (int kk = 0; kk < 4; ++kk) {
      v16bf bfrag = load_frag(&sK[(nt * 16 + l15) * 136 + kk * 32 + khalf]);
      a = __builtin_amdgcn_wmma_f32_16x16x32_bf16(
          false, qf[kk].v, false, bfrag, (short)0, a, false, false);
    }
    s[nt] = a;
  }

  // ---- scale + causal mask ----
  const float scale = 0.08838834764831845f;  // 1/sqrt(128)
#pragma unroll
  for (int nt = 0; nt < 8; ++nt)
#pragma unroll
    for (int e = 0; e < 8; ++e) {
      const int row = m0 + e + hAdd;
      const int col = nt * 16 + l15;
      float sv = s[nt][e] * scale;
      s[nt][e] = (col <= row) ? sv : -1.0e30f;
    }

  // ---- register softmax over the 128-wide row (8 tiles x 16 lanes/half) ----
  float mx[8], sm[8];
#pragma unroll
  for (int e = 0; e < 8; ++e) {
    float m = s[0][e];
#pragma unroll
    for (int nt = 1; nt < 8; ++nt) m = fmaxf(m, s[nt][e]);
#pragma unroll
    for (int off = 1; off < 16; off <<= 1) m = fmaxf(m, __shfl_xor(m, off, 32));
    mx[e] = m;
  }
#pragma unroll
  for (int e = 0; e < 8; ++e) {
    float t = 0.0f;
#pragma unroll
    for (int nt = 0; nt < 8; ++nt) {
      float p = __expf(s[nt][e] - mx[e]);
      s[nt][e] = p;
      t += p;
    }
#pragma unroll
    for (int off = 1; off < 16; off <<= 1) t += __shfl_xor(t, off, 32);
    sm[e] = t;
  }
#pragma unroll
  for (int e = 0; e < 8; ++e) {
    float r = __builtin_amdgcn_rcpf(sm[e]);
#pragma unroll
    for (int nt = 0; nt < 8; ++nt) s[nt][e] *= r;
  }

  // ---- restage att (C-layout regs -> A-layout LDS strip) ----
  __bf16* aw = &sAtt[wave * 16 * 136];
#pragma unroll
  for (int nt = 0; nt < 8; ++nt)
#pragma unroll
    for (int e = 0; e < 8; ++e)
      aw[(e + hAdd) * 136 + nt * 16 + l15] = (__bf16)s[nt][e];
  __syncthreads();

  // ---- out = att @ v ----
  Frag af[4];
#pragma unroll
  for (int kk = 0; kk < 4; ++kk) {
    const __bf16* p = &aw[l15 * 136 + kk * 32 + khalf];
    af[kk].u[0] = *(const uint4*)p;
    af[kk].u[1] = *(const uint4*)(p + 16);
  }
#pragma unroll
  for (int nt = 0; nt < 8; ++nt) {
    v8f o = zero;
#pragma unroll
    for (int kk = 0; kk < 4; ++kk) {
      v16bf bfrag = load_frag(&sVT[(nt * 16 + l15) * 136 + kk * 32 + khalf]);
      o = __builtin_amdgcn_wmma_f32_16x16x32_bf16(
          false, af[kk].v, false, bfrag, (short)0, o, false, false);
    }
#pragma unroll
    for (int e = 0; e < 8; ++e) {
      const size_t row = tok0 + m0 + e + hAdd;
      const int col = h * 128 + nt * 16 + l15;
      y[row * 2048 + col] = (__bf16)o[e];
    }
  }
}

// ---------------------------------------------------------------------------
// Host-side launcher
// ---------------------------------------------------------------------------
extern "C" void kernel_launch(void* const* d_in, const int* in_sizes, int n_in,
                              void* d_out, int out_size, void* d_ws, size_t ws_size,
                              hipStream_t stream) {
  (void)in_sizes; (void)n_in; (void)out_size; (void)ws_size;

  const float* x     = (const float*)d_in[0];   // (4, 8192, 2048)
  const float* Wqkv  = (const float*)d_in[1];   // (2048, 6144)
  const float* Wproj = (const float*)d_in[2];   // (2048, 2048)
  float* out = (float*)d_out;                   // (4, 8192, 2048) fp32

  const int C  = 2048;
  const int N1 = 3 * C;                         // 6144
  const size_t M = (size_t)4 * 8192;            // 32768 tokens

  // Workspace layout (bf16)
  __bf16* xb     = (__bf16*)d_ws;               // M x C
  __bf16* wqkvT  = xb + M * C;                  // N1 x C  (W_qkv transposed)
  __bf16* wprojT = wqkvT + (size_t)N1 * C;      // C x C   (W_proj transposed)
  __bf16* qkvb   = wprojT + (size_t)C * C;      // M x N1
  __bf16* yb     = qkvb + M * N1;               // M x C

  // 1. x -> bf16
  {
    size_t n = M * (size_t)C;
    size_t blocks = (n / 4 + 255) / 256;
    cvt_bf16<<<(unsigned)blocks, 256, 0, stream>>>(x, xb, n);
  }
  // 2. W_qkv (C x N1) -> bf16 transposed (N1 x C)
  transpose_cvt<<<dim3(N1 / 32, C / 32), dim3(32, 8), 0, stream>>>(Wqkv, wqkvT, C, N1);
  // 3. W_proj (C x C) -> bf16 transposed
  transpose_cvt<<<dim3(C / 32, C / 32), dim3(32, 8), 0, stream>>>(Wproj, wprojT, C, C);
  // 4. qkv = x @ W_qkv   (bf16 out)
  gemm_bf16_kernel<true><<<dim3(N1 / 128, (unsigned)(M / 128)), 256, 0, stream>>>(
      xb, wqkvT, (void*)qkvb, (int)M, N1, C);
  // 5. block-causal attention -> y (bf16)
  attn_kernel<<<4 * 64 * 16, 256, 0, stream>>>(qkvb, yb);
  // 6. out = y @ W_proj  (fp32 out)
  gemm_bf16_kernel<false><<<dim3(C / 128, (unsigned)(M / 128)), 256, 0, stream>>>(
      yb, wprojT, (void*)out, (int)M, C, C);
}
